// DynamicalModel_91285234909777
// MI455X (gfx1250) — compile-verified, matrix operand order checked by reading, not run on hardware
//
#include <hip/hip_runtime.h>
#include <hip/hip_bf16.h>

// Problem constants (from reference)
#define BSZ   2048
#define TSTEP 32
#define OUTD  128
#define HD    512
#define NSEGC 1024
#define DIN   384          // 3*OUT
#define G4H   2048         // 4*H
#define KCAT  1408         // [x(384) | pooled(512) | h(512)]
#define KMLP0 896          // [h_n(512) | x(384)]

#define BROW   40            // padded LDS row stride in halfs (80 B = 64 B data + 16 B pad)
#define BROWS  128           // rows per staged weight panel
#define BPANEL (BROWS * BROW)

typedef _Float16 half8   __attribute__((ext_vector_type(8)));
typedef _Float16 half16  __attribute__((ext_vector_type(16)));
typedef float    floatx8 __attribute__((ext_vector_type(8)));

// ---------------------------------------------------------------------------
// v_wmma_f32_16x16x32_f16 operand fragments (wave32).
// 16-bit operand 16x32 layout: lanes 0-15 row r, K=k0+0..7 / k0+16..23;
// lanes 16-31 row r, K=k0+8..15 / k0+24..31.  Two 16-byte loads per lane.
// ---------------------------------------------------------------------------
__device__ __forceinline__ half16 ld_tile16(const _Float16* __restrict__ p) {
  half8 lo = *reinterpret_cast<const half8*>(p);
  half8 hi = *reinterpret_cast<const half8*>(p + 16);
  return __builtin_shufflevector(lo, hi, 0,1,2,3,4,5,6,7,8,9,10,11,12,13,14,15);
}

// Async DMA of a 128-row x 32-K weight slice into LDS buffer `buf`:
// 2 issues/thread x 256 threads x 16 B = 8 KB (row = tid/4 and +64, seg = tid%4).
__device__ __forceinline__ void async_copy_B(const _Float16* __restrict__ W,
                                             _Float16* bsm, int K, int n0, int k0,
                                             int tid, int buf) {
  const int row = tid >> 2;      // 0..63
  const int seg = tid & 3;
  const _Float16* g = W + (size_t)(n0 + row) * K + k0 + seg * 8;
  // Generic pointer into LDS: low 32 bits are the LDS byte offset (LDS aperture).
  uint32_t loff = (uint32_t)(uintptr_t)(bsm + buf * BPANEL + row * BROW + seg * 8);
  asm volatile("global_load_async_to_lds_b128 %0, %1, off"
               :: "v"(loff), "v"((uint64_t)(uintptr_t)g) : "memory");
  const _Float16* g2 = g + (size_t)64 * K;
  uint32_t loff2 = loff + 64 * BROW * 2;
  asm volatile("global_load_async_to_lds_b128 %0, %1, off"
               :: "v"(loff2), "v"((uint64_t)(uintptr_t)g2) : "memory");
}

// 8 WMMAs over one staged 128-row panel slice.  (Letting the scheduler pick
// the B staging registers is the no-spill optimum; LDS latency per fragment
// is hidden by multi-wave occupancy.)
__device__ __forceinline__ void compute_slice(const _Float16* bb, half16 a_cur,
                                              int l15, floatx8 (&acc)[8]) {
#pragma unroll
  for (int s = 0; s < 8; ++s) {
    half16 b = ld_tile16(bb + (s * 16 + l15) * BROW);   // ds_load_b128 x2
    acc[s] = __builtin_amdgcn_wmma_f32_16x16x32_f16(
        false, a_cur, false, b, (short)0, acc[s], false, false);
  }
}

// Block tile: 128(M) x 128(N); wave = 16(M) x 128(N) (8 accumulators).
// Weight panel double-buffered in LDS via ASYNCcnt DMA; A fragment pipelined
// in registers.  Last k-step peeled: branch-free hot loop.
__device__ __forceinline__ void wmma_tile(const _Float16* __restrict__ A,
                                          const _Float16* __restrict__ W,
                                          _Float16* bsm, int K,
                                          int m0, int n0, int lane, int tid,
                                          floatx8 (&acc)[8]) {
  const int l15  = lane & 15;
  const int hsel = (lane & 16) ? 8 : 0;
  const _Float16* __restrict__ arow = A + (size_t)(m0 + l15) * K + hsel;
  const int nk = K >> 5;

  async_copy_B(W, bsm, K, n0, 0, tid, 0);
  half16 a_cur = ld_tile16(arow);

  for (int i = 0; i < nk - 1; ++i) {
    async_copy_B(W, bsm, K, n0, (i + 1) << 5, tid, (i + 1) & 1);
    asm volatile("s_wait_asynccnt 0x2" ::: "memory");   // current slice landed
    __syncthreads();
    half16 a_next = ld_tile16(arow + ((i + 1) << 5));
    __builtin_prefetch(arow + ((i + 1) << 5) + 128, 0, 3);
    compute_slice(bsm + (i & 1) * BPANEL + hsel, a_cur, l15, acc);
    __syncthreads();                                    // before buffer reuse
    a_cur = a_next;
  }

  asm volatile("s_wait_asynccnt 0x0" ::: "memory");
  __syncthreads();
  compute_slice(bsm + ((nk - 1) & 1) * BPANEL + hsel, a_cur, l15, acc);
}

__device__ __forceinline__ float sigm(float x) { return 1.0f / (1.0f + __expf(-x)); }
__device__ __forceinline__ float leaky(float x) { return x > 0.0f ? x : 0.01f * x; }

// ---------------------------------------------------------------------------
// GEMM kernels: blockDim=256 (8 waves), block = 128M x 128N
// ---------------------------------------------------------------------------

// gates = inp @ Wcat^T  (f32 out; biases added in cell kernel)
__global__ void k_gemm_f32(const _Float16* __restrict__ A,
                           const _Float16* __restrict__ W,
                           float* __restrict__ C, int N, int K) {
  __shared__ _Float16 bsm[2 * BPANEL];
  const int lane = threadIdx.x & 31;
  const int wave = threadIdx.x >> 5;
  const int tn = N >> 7;
  const int m0 = (blockIdx.x / tn) * 128 + wave * 16;
  const int n0 = (blockIdx.x % tn) * 128;
  floatx8 acc[8] = {};
  wmma_tile(A, W, bsm, K, m0, n0, lane, threadIdx.x, acc);
  const int l15 = lane & 15;
  const int mh  = (lane & 16) ? 8 : 0;
#pragma unroll
  for (int s = 0; s < 8; ++s)
#pragma unroll
    for (int j = 0; j < 8; ++j)
      C[(size_t)(m0 + mh + j) * N + (n0 + s * 16 + l15)] = acc[s][j];
}

// D = leaky(A @ W^T + bias), f16 out (MLP hidden layers)
__global__ void k_gemm_leaky_f16(const _Float16* __restrict__ A,
                                 const _Float16* __restrict__ W,
                                 const float* __restrict__ bias,
                                 _Float16* __restrict__ D, int N, int K) {
  __shared__ _Float16 bsm[2 * BPANEL];
  const int lane = threadIdx.x & 31;
  const int wave = threadIdx.x >> 5;
  const int tn = N >> 7;
  const int m0 = (blockIdx.x / tn) * 128 + wave * 16;
  const int n0 = (blockIdx.x % tn) * 128;
  floatx8 acc[8] = {};
  wmma_tile(A, W, bsm, K, m0, n0, lane, threadIdx.x, acc);
  const int l15 = lane & 15;
  const int mh  = (lane & 16) ? 8 : 0;
#pragma unroll
  for (int s = 0; s < 8; ++s) {
    const int col = n0 + s * 16 + l15;
    const float bb = bias[col];
#pragma unroll
    for (int j = 0; j < 8; ++j)
      D[(size_t)(m0 + mh + j) * N + col] = (_Float16)leaky(acc[s][j] + bb);
  }
}

// Final layer (N=128, K=1024): deep = A @ W3^T + b3, then pos/vel/acc assembly
__global__ void k_gemm_final(const _Float16* __restrict__ A,
                             const _Float16* __restrict__ W,
                             const float* __restrict__ b3,
                             const float* __restrict__ obs,
                             float* __restrict__ out, int t) {
  __shared__ _Float16 bsm[2 * BPANEL];
  const int lane = threadIdx.x & 31;
  const int wave = threadIdx.x >> 5;
  const int m0 = blockIdx.x * 128 + wave * 16;   // N tiles = 1
  const int n0 = 0;
  floatx8 acc[8] = {};
  wmma_tile(A, W, bsm, 1024, m0, n0, lane, threadIdx.x, acc);
  const int l15 = lane & 15;
  const int mh  = (lane & 16) ? 8 : 0;
#pragma unroll
  for (int s = 0; s < 8; ++s) {
    const int col = n0 + s * 16 + l15;      // 0..127
    const float bb = b3[col];
#pragma unroll
    for (int j = 0; j < 8; ++j) {
      const int row = m0 + mh + j;          // batch index
      const size_t base = ((size_t)row * TSTEP + t) * DIN;
      const float deep = acc[s][j] + bb;
      const float x  = obs[base + col];
      const float vv = obs[base + OUTD + col];
      const float aa = obs[base + 2 * OUTD + col];
      const float vel = vv + 0.5f * (aa + deep);
      const float pos = x + vel + 0.5f * deep;
      out[base + col]            = pos;
      out[base + OUTD + col]     = vel;
      out[base + 2 * OUTD + col] = deep;
    }
  }
}

// ---------------------------------------------------------------------------
// Elementwise / setup kernels
// ---------------------------------------------------------------------------
__global__ void k_zero_f32(float* p, int n) {
  int i = blockIdx.x * blockDim.x + threadIdx.x;
  if (i < n) p[i] = 0.0f;
}

__global__ void k_f32_to_f16(const float* __restrict__ s, _Float16* __restrict__ d, int n) {
  int i = blockIdx.x * blockDim.x + threadIdx.x;
  if (i < n) d[i] = (_Float16)s[i];
}

// Wcat = [W_ih (2048x896) | W_hh (2048x512)] row-major f16, K=1408
__global__ void k_build_wcat(const float* __restrict__ wih, const float* __restrict__ whh,
                             _Float16* __restrict__ wc) {
  int i = blockIdx.x * blockDim.x + threadIdx.x;
  if (i >= G4H * KCAT) return;
  int n = i / KCAT, k = i % KCAT;
  float v = (k < 896) ? wih[(size_t)n * 896 + k] : whh[(size_t)n * 512 + (k - 896)];
  wc[i] = (_Float16)v;
}

// segment boundaries from sorted ids (one pass, time-invariant)
__global__ void k_seg_bounds(const int* __restrict__ ids,
                             int* __restrict__ sstart, int* __restrict__ send) {
  int r = blockIdx.x * blockDim.x + threadIdx.x;
  if (r >= BSZ) return;
  int id = ids[r];
  if (r == 0 || ids[r - 1] != id) sstart[id] = r;
  if (r == BSZ - 1 || ids[r + 1] != id) send[id] = r + 1;
}

// inp[b, :] = [x_t(384) | segment_max(h)(512) | h(512)] as f16
__global__ void k_build_inp(const float* __restrict__ obs, const float* __restrict__ h,
                            const int* __restrict__ ids,
                            const int* __restrict__ sstart, const int* __restrict__ send,
                            _Float16* __restrict__ inp, int t) {
  int i = blockIdx.x * blockDim.x + threadIdx.x;
  if (i >= BSZ * KCAT) return;
  int b = i / KCAT, col = i % KCAT;
  float v;
  if (col < DIN) {
    v = obs[((size_t)b * TSTEP + t) * DIN + col];
  } else if (col < DIN + HD) {
    int f = col - DIN;
    int id = ids[b];
    int s = sstart[id], e = send[id];
    float m = -INFINITY;
    for (int r = s; r < e; ++r) m = fmaxf(m, h[(size_t)r * HD + f]);
    v = m;
  } else {
    v = h[(size_t)b * HD + (col - DIN - HD)];
  }
  inp[i] = (_Float16)v;
}

// LSTM cell + build mlp_in = [h_n(512) | x_t(384)] f16
__global__ void k_lstm_cell(const float* __restrict__ gates,
                            const float* __restrict__ bih, const float* __restrict__ bhh,
                            float* __restrict__ c, float* __restrict__ h,
                            const float* __restrict__ obs,
                            _Float16* __restrict__ mlp_in, int t) {
  int i = blockIdx.x * blockDim.x + threadIdx.x;
  if (i >= BSZ * KMLP0) return;
  int b = i / KMLP0, j = i % KMLP0;
  if (j < HD) {
    const size_t gb = (size_t)b * G4H;
    float ig = gates[gb + j]            + bih[j]            + bhh[j];
    float fg = gates[gb + HD + j]       + bih[HD + j]       + bhh[HD + j];
    float gg = gates[gb + 2 * HD + j]   + bih[2 * HD + j]   + bhh[2 * HD + j];
    float og = gates[gb + 3 * HD + j]   + bih[3 * HD + j]   + bhh[3 * HD + j];
    float co = c[(size_t)b * HD + j];
    float cn = sigm(fg) * co + sigm(ig) * tanhf(gg);
    float hn = sigm(og) * tanhf(cn);
    c[(size_t)b * HD + j] = cn;
    h[(size_t)b * HD + j] = hn;
    mlp_in[(size_t)b * KMLP0 + j] = (_Float16)hn;
  } else {
    mlp_in[(size_t)b * KMLP0 + j] =
        (_Float16)obs[((size_t)b * TSTEP + t) * DIN + (j - HD)];
  }
}

// ---------------------------------------------------------------------------
// Launch
// ---------------------------------------------------------------------------
static inline int gridFor(long long n, int bs = 256) { return (int)((n + bs - 1) / bs); }

extern "C" void kernel_launch(void* const* d_in, const int* in_sizes, int n_in,
                              void* d_out, int out_size, void* d_ws, size_t ws_size,
                              hipStream_t stream) {
  const float* obs = (const float*)d_in[0];
  const float* wih = (const float*)d_in[1];
  const float* whh = (const float*)d_in[2];
  const float* bih = (const float*)d_in[3];
  const float* bhh = (const float*)d_in[4];
  const float* W0  = (const float*)d_in[5];
  const float* b0  = (const float*)d_in[6];
  const float* W1  = (const float*)d_in[7];
  const float* b1  = (const float*)d_in[8];
  const float* W2  = (const float*)d_in[9];
  const float* b2  = (const float*)d_in[10];
  const float* W3  = (const float*)d_in[11];
  const float* b3  = (const float*)d_in[12];
  const int*   ids = (const int*)d_in[13];
  float* out = (float*)d_out;

  // workspace carve-out (256B aligned)
  uint8_t* wp = (uint8_t*)d_ws;
  auto alloc = [&](size_t bytes) -> void* {
    void* p = wp;
    wp += (bytes + 255) & ~(size_t)255;
    return p;
  };
  float*    h      = (float*)alloc((size_t)BSZ * HD * 4);
  float*    c      = (float*)alloc((size_t)BSZ * HD * 4);
  float*    gates  = (float*)alloc((size_t)BSZ * G4H * 4);
  _Float16* wcat   = (_Float16*)alloc((size_t)G4H * KCAT * 2);
  _Float16* w0h    = (_Float16*)alloc((size_t)512 * 896 * 2);
  _Float16* w1h    = (_Float16*)alloc((size_t)1024 * 512 * 2);
  _Float16* w2h    = (_Float16*)alloc((size_t)1024 * 1024 * 2);
  _Float16* w3h    = (_Float16*)alloc((size_t)OUTD * 1024 * 2);
  _Float16* inp    = (_Float16*)alloc((size_t)BSZ * KCAT * 2);
  _Float16* mlp_in = (_Float16*)alloc((size_t)BSZ * KMLP0 * 2);
  _Float16* act0   = (_Float16*)alloc((size_t)BSZ * 512 * 2);
  _Float16* act1   = (_Float16*)alloc((size_t)BSZ * 1024 * 2);
  _Float16* act2   = (_Float16*)alloc((size_t)BSZ * 1024 * 2);
  int*      sstart = (int*)alloc((size_t)NSEGC * 4);
  int*      send   = (int*)alloc((size_t)NSEGC * 4);

  // one-time (per call) setup: zero state, seg bounds, f16 weight packs
  k_zero_f32<<<gridFor((long long)BSZ * HD), 256, 0, stream>>>(h, BSZ * HD);
  k_zero_f32<<<gridFor((long long)BSZ * HD), 256, 0, stream>>>(c, BSZ * HD);
  k_seg_bounds<<<gridFor(BSZ), 256, 0, stream>>>(ids, sstart, send);
  k_build_wcat<<<gridFor((long long)G4H * KCAT), 256, 0, stream>>>(wih, whh, wcat);
  k_f32_to_f16<<<gridFor(512 * 896), 256, 0, stream>>>(W0, w0h, 512 * 896);
  k_f32_to_f16<<<gridFor(1024 * 512), 256, 0, stream>>>(W1, w1h, 1024 * 512);
  k_f32_to_f16<<<gridFor(1024 * 1024), 256, 0, stream>>>(W2, w2h, 1024 * 1024);
  k_f32_to_f16<<<gridFor(OUTD * 1024), 256, 0, stream>>>(W3, w3h, OUTD * 1024);

  // blocks = (M/128) * (N/128)
  const int g_gates = (BSZ / 128) * (G4H / 128);   // 256
  const int g_l0    = (BSZ / 128) * (512 / 128);   // 64
  const int g_l1    = (BSZ / 128) * (1024 / 128);  // 128
  const int g_l3    = (BSZ / 128) * (OUTD / 128);  // 16

  for (int t = 0; t < TSTEP; ++t) {
    k_build_inp<<<gridFor((long long)BSZ * KCAT), 256, 0, stream>>>(
        obs, h, ids, sstart, send, inp, t);
    k_gemm_f32<<<g_gates, 256, 0, stream>>>(inp, wcat, gates, G4H, KCAT);
    k_lstm_cell<<<gridFor((long long)BSZ * KMLP0), 256, 0, stream>>>(
        gates, bih, bhh, c, h, obs, mlp_in, t);
    k_gemm_leaky_f16<<<g_l0, 256, 0, stream>>>(mlp_in, w0h, b0, act0, 512, KMLP0);
    k_gemm_leaky_f16<<<g_l1, 256, 0, stream>>>(act0, w1h, b1, act1, 1024, 512);
    k_gemm_leaky_f16<<<g_l1, 256, 0, stream>>>(act1, w2h, b2, act2, 1024, 1024);
    k_gemm_final<<<g_l3, 256, 0, stream>>>(act2, w3h, b3, obs, out, t);
  }
}
